// RNNStateEncoder_67860483277525
// MI455X (gfx1250) — matrix operand order repaired and checked by reading, not compiled, not run on hardware
//
#include <hip/hip_runtime.h>

typedef __attribute__((ext_vector_type(16))) __bf16 v16bf;
typedef __attribute__((ext_vector_type(8)))  __bf16 v8bf;
typedef __attribute__((ext_vector_type(8)))  float  v8f;

#define TT 128
#define NE 512     // envs
#define DD 512     // input dim
#define HH 512     // hidden dim
#define G3 1536    // 3*H
#define EB 32      // env blocks of 16
#define NT 96      // 16-col tiles across 3H
#define LSTR 520   // padded LDS row stride in bf16 (16B-aligned, breaks bank conflicts)

__device__ __forceinline__ float sigm(float x) { return 1.0f / (1.0f + __expf(-x)); }

__global__ void rnn_f32_to_bf16(const float* __restrict__ in, __bf16* __restrict__ out, int n) {
  int i = blockIdx.x * blockDim.x + threadIdx.x;
  if (i < n) out[i] = (__bf16)in[i];
}

// gates_x = x @ w_ih^T + b_ih ; stored bf16 in D-fragment order [mt][nt][lane][8]
__global__ __launch_bounds__(256) void rnn_gatesx_gemm(
    const __bf16* __restrict__ xb, const __bf16* __restrict__ wih,
    const float* __restrict__ bih, __bf16* __restrict__ gxf) {
  const int lane = threadIdx.x & 31;
  const int wave = threadIdx.x >> 5;
  const int l16 = lane & 15, half = lane >> 4;
  const int mt = blockIdx.x;                 // 0..4095  (= t*32 + env_block)
  const int nt = blockIdx.y * 8 + wave;      // 0..95
  const __bf16* arow = xb  + (size_t)(mt * 16 + l16) * DD;
  const __bf16* brow = wih + (size_t)(nt * 16 + l16) * DD;
  v8f acc = {};
  for (int kk = 0; kk < DD; kk += 32) {
    v8bf alo = *(const v8bf*)(arow + kk + half * 8);
    v8bf ahi = *(const v8bf*)(arow + kk + 16 + half * 8);
    v16bf a, b;
#pragma unroll
    for (int i = 0; i < 8; ++i) { a[i] = alo[i]; a[i + 8] = ahi[i]; }
    v8bf blo = *(const v8bf*)(brow + kk + half * 8);
    v8bf bhi = *(const v8bf*)(brow + kk + 16 + half * 8);
#pragma unroll
    for (int i = 0; i < 8; ++i) { b[i] = blo[i]; b[i + 8] = bhi[i]; }
    acc = __builtin_amdgcn_wmma_f32_16x16x32_bf16(false, a, false, b, (short)0, acc, false, false);
  }
  float bi = bih[nt * 16 + l16];
  v8bf st;
#pragma unroll
  for (int r = 0; r < 8; ++r) st[r] = (__bf16)(acc[r] + bi);
  *(v8bf*)(gxf + (((size_t)mt * NT + nt) * 32 + lane) * 8) = st;
}

// Sequential masked GRU scan. 1 block = 16 envs, h kept in registers (D-frag order).
__global__ __launch_bounds__(256) void rnn_gru_scan(
    const __bf16* __restrict__ gxf, const __bf16* __restrict__ whh,
    const float* __restrict__ bhh, const float* __restrict__ h0,
    const float* __restrict__ masks, float* __restrict__ out) {
  const int lane = threadIdx.x & 31;
  const int wave = threadIdx.x >> 5;
  const int l16 = lane & 15, half = lane >> 4;
  const int eb = blockIdx.x;

  __shared__ __align__(32) __bf16 hlds[16 * LSTR];

  float hreg[4][8];
  float bh[4][3];
#pragma unroll
  for (int hv = 0; hv < 4; ++hv) {
    const int j = (wave * 4 + hv) * 16 + l16;
#pragma unroll
    for (int g = 0; g < 3; ++g) bh[hv][g] = bhh[g * HH + j];
#pragma unroll
    for (int r = 0; r < 8; ++r) {
      const int m = r + 8 * half;
      hreg[hv][r] = h0[(size_t)(eb * 16 + m) * HH + j];
    }
  }

  for (int t = 0; t < TT; ++t) {
    // ---- phase A: apply episode-reset mask, publish bf16 h to LDS ----
    float mv[8];
#pragma unroll
    for (int r = 0; r < 8; ++r)
      mv[r] = masks[(size_t)t * NE + eb * 16 + r + 8 * half];
#pragma unroll
    for (int hv = 0; hv < 4; ++hv) {
      const int j = (wave * 4 + hv) * 16 + l16;
#pragma unroll
      for (int r = 0; r < 8; ++r) {
        const int m = r + 8 * half;
        const float hm = hreg[hv][r] * mv[r];
        hreg[hv][r] = hm;
        hlds[m * LSTR + j] = (__bf16)hm;
      }
    }
    __syncthreads();

    // ---- phase B: gh = h @ w_hh^T via WMMA (A from LDS, B from L2) ----
    v8f acc[4][3];
#pragma unroll
    for (int hv = 0; hv < 4; ++hv)
#pragma unroll
      for (int g = 0; g < 3; ++g) acc[hv][g] = (v8f){};
    for (int kk = 0; kk < HH; kk += 32) {
      v8bf alo = *(const v8bf*)(&hlds[l16 * LSTR + kk + half * 8]);
      v8bf ahi = *(const v8bf*)(&hlds[l16 * LSTR + kk + 16 + half * 8]);
      v16bf a;
#pragma unroll
      for (int i = 0; i < 8; ++i) { a[i] = alo[i]; a[i + 8] = ahi[i]; }
#pragma unroll
      for (int hv = 0; hv < 4; ++hv) {
#pragma unroll
        for (int g = 0; g < 3; ++g) {
          const int j = g * HH + (wave * 4 + hv) * 16 + l16;
          v8bf blo = *(const v8bf*)(whh + (size_t)j * HH + kk + half * 8);
          v8bf bhi = *(const v8bf*)(whh + (size_t)j * HH + kk + 16 + half * 8);
          v16bf b;
#pragma unroll
          for (int i = 0; i < 8; ++i) { b[i] = blo[i]; b[i + 8] = bhi[i]; }
          acc[hv][g] = __builtin_amdgcn_wmma_f32_16x16x32_bf16(
              false, a, false, b, (short)0, acc[hv][g], false, false);
        }
      }
    }

    // ---- phase C: gate math + state update + output ----
    const size_t mt = (size_t)t * EB + eb;
#pragma unroll
    for (int hv = 0; hv < 4; ++hv) {
      const int ht = wave * 4 + hv;
      const int j = ht * 16 + l16;
      v8bf gr = *(const v8bf*)(gxf + ((mt * NT + (0 * 32 + ht)) * 32 + lane) * 8);
      v8bf gz = *(const v8bf*)(gxf + ((mt * NT + (1 * 32 + ht)) * 32 + lane) * 8);
      v8bf gn = *(const v8bf*)(gxf + ((mt * NT + (2 * 32 + ht)) * 32 + lane) * 8);
#pragma unroll
      for (int r = 0; r < 8; ++r) {
        const int m = r + 8 * half;
        const float rg = sigm((float)gr[r] + acc[hv][0][r] + bh[hv][0]);
        const float zg = sigm((float)gz[r] + acc[hv][1][r] + bh[hv][1]);
        const float ng = tanhf((float)gn[r] + rg * (acc[hv][2][r] + bh[hv][2]));
        const float hn = (1.0f - zg) * ng + zg * hreg[hv][r];
        hreg[hv][r] = hn;
        out[((size_t)t * NE + eb * 16 + m) * HH + j] = hn;
        if (t == TT - 1)
          out[(size_t)TT * NE * HH + (size_t)(eb * 16 + m) * HH + j] = hn;
      }
    }
    __syncthreads();
  }
}

extern "C" void kernel_launch(void* const* d_in, const int* in_sizes, int n_in,
                              void* d_out, int out_size, void* d_ws, size_t ws_size,
                              hipStream_t stream) {
  const float* x   = (const float*)d_in[0];
  const float* h0  = (const float*)d_in[1];
  const float* ms  = (const float*)d_in[2];
  const float* wih = (const float*)d_in[3];
  const float* whh = (const float*)d_in[4];
  const float* bih = (const float*)d_in[5];
  const float* bhh = (const float*)d_in[6];
  float* out = (float*)d_out;

  char* ws = (char*)d_ws;
  __bf16* xb   = (__bf16*)ws;                                  // T*N*D bf16 = 64 MB
  __bf16* wihb = (__bf16*)(ws + (size_t)TT * NE * DD * 2);     // 1.5 MB
  __bf16* whhb = wihb + (size_t)G3 * DD;                       // 1.5 MB
  __bf16* gxf  = whhb + (size_t)G3 * HH;                       // T*N*3H bf16 = 192 MB

  const int nx = TT * NE * DD;
  rnn_f32_to_bf16<<<(nx + 255) / 256, 256, 0, stream>>>(x, xb, nx);
  rnn_f32_to_bf16<<<(G3 * DD + 255) / 256, 256, 0, stream>>>(wih, wihb, G3 * DD);
  rnn_f32_to_bf16<<<(G3 * HH + 255) / 256, 256, 0, stream>>>(whh, whhb, G3 * HH);

  dim3 g1(TT * NE / 16, NT / 8);  // 4096 x 12, 8 waves/block, 1 D-tile per wave
  rnn_gatesx_gemm<<<g1, 256, 0, stream>>>(xb, wihb, bih, gxf);

  rnn_gru_scan<<<EB, 256, 0, stream>>>(gxf, whhb, bhh, h0, ms, out);
}
